// multiHeadAttention_44959717654957
// MI455X (gfx1250) — compile-verified, matrix operand order checked by reading, not run on hardware
//
#include <hip/hip_runtime.h>
#include <hip/hip_bf16.h>

typedef __attribute__((ext_vector_type(16))) _Float16 v16h;
typedef __attribute__((ext_vector_type(8)))  float    v8f;

#define B_  2
#define S_  2048
#define D_  1024
#define H_  16
#define DK_ 64

static __device__ __forceinline__ v8f wmma_f16(v16h a, v16h b, v8f c) {
  // D = A(16x32 f16) x B(32x16 f16) + C(16x16 f32)
  return __builtin_amdgcn_wmma_f32_16x16x32_f16(false, a, false, b, (short)0, c,
                                                false, false);
}

// ---------------------------------------------------------------------------
// Q/K/V projection: Y = X @ W^T + b, output stored f16 head-major.
//   transposeOut == 0 : out[b][h][s][d]   (Q, K)
//   transposeOut == 1 : out[b][h][d][s]   (V -> PV B-fragments contiguous)
// Block = 256 threads (8 waves); each wave owns a 32x64 output strip:
// per k-step 2 A-frags + 4 B-frags feed 8 WMMAs (0.75 frag-loads / WMMA).
// ---------------------------------------------------------------------------
__global__ void mha_proj_f16(const float* __restrict__ X, const float* __restrict__ W,
                             const float* __restrict__ bias, _Float16* __restrict__ out,
                             int transposeOut) {
  const int lane = threadIdx.x & 31;
  const int wave = threadIdx.x >> 5;
  const int m    = lane & 15;
  const int half = lane >> 4;
  const int row0 = (blockIdx.x * 8 + wave) * 32;  // token rows
  const int col0 = blockIdx.y * 64;               // output features

  const float* a0p = X + (size_t)(row0 + m) * D_ + half * 16;
  const float* a1p = X + (size_t)(row0 + 16 + m) * D_ + half * 16;
  const float* b0p = W + (size_t)(col0 + m) * D_ + half * 16;
  const float* b1p = W + (size_t)(col0 + 16 + m) * D_ + half * 16;
  const float* b2p = W + (size_t)(col0 + 32 + m) * D_ + half * 16;
  const float* b3p = W + (size_t)(col0 + 48 + m) * D_ + half * 16;

  v8f acc[2][4] = {};
  for (int k0 = 0; k0 < D_; k0 += 32) {
    v16h a0, a1, b0, b1, b2, b3;
#pragma unroll
    for (int i = 0; i < 16; ++i) {
      a0[i] = (_Float16)a0p[k0 + i];
      a1[i] = (_Float16)a1p[k0 + i];
      b0[i] = (_Float16)b0p[k0 + i];
      b1[i] = (_Float16)b1p[k0 + i];
      b2[i] = (_Float16)b2p[k0 + i];
      b3[i] = (_Float16)b3p[k0 + i];
    }
    acc[0][0] = wmma_f16(a0, b0, acc[0][0]);
    acc[0][1] = wmma_f16(a0, b1, acc[0][1]);
    acc[0][2] = wmma_f16(a0, b2, acc[0][2]);
    acc[0][3] = wmma_f16(a0, b3, acc[0][3]);
    acc[1][0] = wmma_f16(a1, b0, acc[1][0]);
    acc[1][1] = wmma_f16(a1, b1, acc[1][1]);
    acc[1][2] = wmma_f16(a1, b2, acc[1][2]);
    acc[1][3] = wmma_f16(a1, b3, acc[1][3]);
  }

#pragma unroll
  for (int rt = 0; rt < 2; ++rt) {
#pragma unroll
    for (int c = 0; c < 4; ++c) {
      const int   f  = col0 + c * 16 + m;   // output feature (C-layout N)
      const int   hh = f >> 6;
      const int   d  = f & 63;
      const float bv = bias[f];
#pragma unroll
      for (int r = 0; r < 8; ++r) {
        const int t  = row0 + rt * 16 + r + 8 * half;  // token (C-layout M)
        const int bb = t >> 11;
        const int s  = t & (S_ - 1);
        const float val = acc[rt][c][r] + bv;
        const size_t idx = transposeOut
            ? ((((size_t)bb * H_ + hh) * DK_ + d) * S_ + s)
            : ((((size_t)bb * H_ + hh) * S_ + s) * DK_ + d);
        out[idx] = (_Float16)val;
      }
    }
  }
}

// ---------------------------------------------------------------------------
// Flash attention: one wave per 32-query tile per (b,h).
// Per 32-key step: 8 WMMAs QK^T + 8 WMMAs P.V from only 4 K-frags + 4 V-frags.
// Online softmax in exp2 domain (scale = log2(e)/sqrt(dk) folded into one mul).
// Row stats live in lanes (row = lane%16, halves combined via shfl_xor 16);
// score -> P transpose goes through LDS (single-wave workgroup: barriers~free).
// ---------------------------------------------------------------------------
__global__ void mha_flash_attn(const _Float16* __restrict__ qh,
                               const _Float16* __restrict__ kh,
                               const _Float16* __restrict__ vT,
                               const int* __restrict__ mask,
                               _Float16* __restrict__ xh) {
  __shared__ __align__(32) float    sc[2][16][32];
  __shared__ __align__(32) _Float16 pbuf[2][16][32];
  __shared__ float alph[2][16];
  __shared__ float lrow[2][16];

  const int lane = threadIdx.x;   // block = 32 threads = 1 wave
  const int m    = lane & 15;
  const int half = lane >> 4;
  const int q0   = blockIdx.x * 32;
  const int h    = blockIdx.y;
  const int b    = blockIdx.z;

  const _Float16* qbase = qh + ((size_t)b * H_ + h) * S_ * DK_;
  const _Float16* kbase = kh + ((size_t)b * H_ + h) * S_ * DK_;
  const _Float16* vbase = vT + ((size_t)b * H_ + h) * DK_ * S_;
  const int* mrows[2];
  mrows[0] = mask + ((size_t)b * S_ + q0 + m) * S_ + half * 16;
  mrows[1] = mrows[0] + (size_t)16 * S_;

  // Q A-fragments, resident for the whole pass (2 q-tiles x 2 dk-chunks)
  const v16h qa00 = *(const v16h*)(qbase + (size_t)(q0 + m) * DK_ + half * 16);
  const v16h qa01 = *(const v16h*)(qbase + (size_t)(q0 + m) * DK_ + 32 + half * 16);
  const v16h qa10 = *(const v16h*)(qbase + (size_t)(q0 + 16 + m) * DK_ + half * 16);
  const v16h qa11 = *(const v16h*)(qbase + (size_t)(q0 + 16 + m) * DK_ + 32 + half * 16);

  float mi[2] = {-3.0e38f, -3.0e38f};
  float li[2] = {0.0f, 0.0f};
  v8f acc[2][4] = {};

  const float SCL = 0.125f * 1.4426950408889634f;  // log2(e)/sqrt(dk)

  for (int j0 = 0; j0 < S_; j0 += 32) {
    // ---- scores: four 16x16 tiles (2 q-tiles x 2 key-tiles) ----
    const _Float16* kp0 = kbase + (size_t)(j0 + m) * DK_;
    const _Float16* kp1 = kbase + (size_t)(j0 + 16 + m) * DK_;
    const v16h kb00 = *(const v16h*)(kp0 + half * 16);
    const v16h kb01 = *(const v16h*)(kp0 + 32 + half * 16);
    const v16h kb10 = *(const v16h*)(kp1 + half * 16);
    const v16h kb11 = *(const v16h*)(kp1 + 32 + half * 16);

    if (j0 + 32 < S_) {
      __builtin_prefetch(kbase + (size_t)(j0 + 32 + m) * DK_, 0, 0);
      __builtin_prefetch(vbase + (size_t)m * S_ + j0 + 32, 0, 0);
    }

    v8f s00 = {}, s01 = {}, s10 = {}, s11 = {};
    s00 = wmma_f16(qa00, kb00, s00);
    s00 = wmma_f16(qa01, kb01, s00);
    s01 = wmma_f16(qa00, kb10, s01);
    s01 = wmma_f16(qa01, kb11, s01);
    s10 = wmma_f16(qa10, kb00, s10);
    s10 = wmma_f16(qa11, kb01, s10);
    s11 = wmma_f16(qa10, kb10, s11);
    s11 = wmma_f16(qa11, kb11, s11);

#pragma unroll
    for (int r = 0; r < 8; ++r) {
      sc[0][r + 8 * half][m]      = s00[r] * SCL;
      sc[0][r + 8 * half][16 + m] = s01[r] * SCL;
      sc[1][r + 8 * half][m]      = s10[r] * SCL;
      sc[1][r + 8 * half][16 + m] = s11[r] * SCL;
    }
    __syncthreads();

    // ---- online softmax (exp2 domain): lane -> row m, cols [16h,16h+16) ----
#pragma unroll
    for (int t = 0; t < 2; ++t) {
      const int* mp = mrows[t] + j0;
      float sv[16];
      float tmax = -3.0e38f;
#pragma unroll
      for (int i = 0; i < 16; ++i) {
        float x = sc[t][m][half * 16 + i];
        if (mp[i] == 0) x = -1.0e9f;
        sv[i] = x;
        tmax = fmaxf(tmax, x);
      }
      tmax = fmaxf(tmax, __shfl_xor(tmax, 16));
      const float newm  = fmaxf(mi[t], tmax);
      const float alpha = exp2f(mi[t] - newm);
      float psum = 0.0f;
#pragma unroll
      for (int i = 0; i < 16; ++i) {
        const float p = exp2f(sv[i] - newm);
        psum += p;
        pbuf[t][m][half * 16 + i] = (_Float16)p;
      }
      psum += __shfl_xor(psum, 16);
      li[t] = li[t] * alpha + psum;
      mi[t] = newm;
      if (half == 0) alph[t][m] = alpha;
    }
    __syncthreads();

    // ---- rescale accumulators, then acc += P x V (V frags shared) ----
    const _Float16* vp = vbase + (size_t)m * S_ + j0 + half * 16;  // vT[d][s]
    const v16h vb0 = *(const v16h*)(vp);
    const v16h vb1 = *(const v16h*)(vp + 16 * S_);
    const v16h vb2 = *(const v16h*)(vp + 32 * S_);
    const v16h vb3 = *(const v16h*)(vp + 48 * S_);

#pragma unroll
    for (int t = 0; t < 2; ++t) {
      const v16h pa = *(const v16h*)(&pbuf[t][m][half * 16]);
#pragma unroll
      for (int r = 0; r < 8; ++r) {
        const float a_ = alph[t][r + 8 * half];
        acc[t][0][r] *= a_; acc[t][1][r] *= a_;
        acc[t][2][r] *= a_; acc[t][3][r] *= a_;
      }
      acc[t][0] = wmma_f16(pa, vb0, acc[t][0]);
      acc[t][1] = wmma_f16(pa, vb1, acc[t][1]);
      acc[t][2] = wmma_f16(pa, vb2, acc[t][2]);
      acc[t][3] = wmma_f16(pa, vb3, acc[t][3]);
    }
    __syncthreads();
  }

  if (half == 0) {
    lrow[0][m] = li[0];
    lrow[1][m] = li[1];
  }
  __syncthreads();

#pragma unroll
  for (int t = 0; t < 2; ++t) {
#pragma unroll
    for (int r = 0; r < 8; ++r) {
      const int M = r + 8 * half;
      const float inv = 1.0f / lrow[t][M];
      _Float16* xr = xh + ((size_t)b * S_ + q0 + t * 16 + M) * D_ + h * DK_ + m;
      xr[0]  = (_Float16)(acc[t][0][r] * inv);
      xr[16] = (_Float16)(acc[t][1][r] * inv);
      xr[32] = (_Float16)(acc[t][2][r] * inv);
      xr[48] = (_Float16)(acc[t][3][r] * inv);
    }
  }
}

// ---------------------------------------------------------------------------
// Output projection: out = Xh @ W_o^T + b_o  (Xh f16, out fp32), 32x64 strips.
// ---------------------------------------------------------------------------
__global__ void mha_out_proj(const _Float16* __restrict__ Xh, const float* __restrict__ W,
                             const float* __restrict__ bias, float* __restrict__ out) {
  const int lane = threadIdx.x & 31;
  const int wave = threadIdx.x >> 5;
  const int m    = lane & 15;
  const int half = lane >> 4;
  const int row0 = (blockIdx.x * 8 + wave) * 32;
  const int col0 = blockIdx.y * 64;

  const _Float16* a0p = Xh + (size_t)(row0 + m) * D_ + half * 16;
  const _Float16* a1p = Xh + (size_t)(row0 + 16 + m) * D_ + half * 16;
  const float* b0p = W + (size_t)(col0 + m) * D_ + half * 16;
  const float* b1p = W + (size_t)(col0 + 16 + m) * D_ + half * 16;
  const float* b2p = W + (size_t)(col0 + 32 + m) * D_ + half * 16;
  const float* b3p = W + (size_t)(col0 + 48 + m) * D_ + half * 16;

  v8f acc[2][4] = {};
  for (int k0 = 0; k0 < D_; k0 += 32) {
    const v16h a0 = *(const v16h*)(a0p + k0);
    const v16h a1 = *(const v16h*)(a1p + k0);
    v16h b0, b1, b2, b3;
#pragma unroll
    for (int i = 0; i < 16; ++i) {
      b0[i] = (_Float16)b0p[k0 + i];
      b1[i] = (_Float16)b1p[k0 + i];
      b2[i] = (_Float16)b2p[k0 + i];
      b3[i] = (_Float16)b3p[k0 + i];
    }
    acc[0][0] = wmma_f16(a0, b0, acc[0][0]);
    acc[0][1] = wmma_f16(a0, b1, acc[0][1]);
    acc[0][2] = wmma_f16(a0, b2, acc[0][2]);
    acc[0][3] = wmma_f16(a0, b3, acc[0][3]);
    acc[1][0] = wmma_f16(a1, b0, acc[1][0]);
    acc[1][1] = wmma_f16(a1, b1, acc[1][1]);
    acc[1][2] = wmma_f16(a1, b2, acc[1][2]);
    acc[1][3] = wmma_f16(a1, b3, acc[1][3]);
  }

#pragma unroll
  for (int rt = 0; rt < 2; ++rt) {
#pragma unroll
    for (int c = 0; c < 4; ++c) {
      const float bv = bias[col0 + c * 16 + m];
#pragma unroll
      for (int r = 0; r < 8; ++r) {
        out[(size_t)(row0 + rt * 16 + r + 8 * half) * D_ + col0 + c * 16 + m] =
            acc[rt][c][r] + bv;
      }
    }
  }
}

// ---------------------------------------------------------------------------
extern "C" void kernel_launch(void* const* d_in, const int* in_sizes, int n_in,
                              void* d_out, int out_size, void* d_ws, size_t ws_size,
                              hipStream_t stream) {
  const float* query = (const float*)d_in[0];
  const float* key   = (const float*)d_in[1];
  const float* value = (const float*)d_in[2];
  const int*   mask  = (const int*)d_in[3];
  const float* w_q   = (const float*)d_in[4];
  const float* b_q   = (const float*)d_in[5];
  const float* w_k   = (const float*)d_in[6];
  const float* b_k   = (const float*)d_in[7];
  const float* w_v   = (const float*)d_in[8];
  const float* b_v   = (const float*)d_in[9];
  const float* w_o   = (const float*)d_in[10];
  const float* b_o   = (const float*)d_in[11];
  float* out = (float*)d_out;

  const size_t QKV = (size_t)B_ * H_ * S_ * DK_;   // 4,194,304 elements
  _Float16* qh = (_Float16*)d_ws;                  // [B,H,S,dk]
  _Float16* kh = qh + QKV;                         // [B,H,S,dk]
  _Float16* vT = kh + QKV;                         // [B,H,dk,S]
  _Float16* xh = vT + QKV;                         // [B,S,D]

  const dim3 gblk(256);
  const dim3 ggrid((B_ * S_) / (32 * 8), D_ / 64);  // 16 x 16 strip grid

  mha_proj_f16<<<ggrid, gblk, 0, stream>>>(query, w_q, b_q, qh, 0);
  mha_proj_f16<<<ggrid, gblk, 0, stream>>>(key,   w_k, b_k, kh, 0);
  mha_proj_f16<<<ggrid, gblk, 0, stream>>>(value, w_v, b_v, vT, 1);

  mha_flash_attn<<<dim3(S_ / 32, H_, B_), 32, 0, stream>>>(qh, kh, vT, mask, xh);

  mha_out_proj<<<ggrid, gblk, 0, stream>>>(xh, w_o, b_o, out);
}
